// Wmamba_57251914055771
// MI455X (gfx1250) — matrix operand order repaired and checked by reading, not compiled, not run on hardware
//
#include <hip/hip_runtime.h>
#include <hip/hip_bf16.h>

// ---------------------------------------------------------------------------
// Windowed Mamba block, fused per-window kernel for gfx1250 (MI455X).
// All three GEMMs (tn@W_in 64x128x512, xc@W_x 64x256x48(40), y@W_out
// 64x256x128) run on the matrix pipe via v_wmma_f32_16x16x32_bf16.
// Weights pre-packed to bf16 WMMA-B fragment order in d_ws.
// LDS = 128 KB/workgroup -> 2 workgroups per 320 KB WGP (16 waves resident).
// ---------------------------------------------------------------------------

typedef __attribute__((ext_vector_type(16))) __bf16        v16bf;
typedef __attribute__((ext_vector_type(8)))  float         v8f;
typedef __attribute__((ext_vector_type(4)))  unsigned int  v4u;

union BFrag {
    v16bf v;
    v4u   u4[2];
    unsigned int u[8];
    unsigned short us[16];
};

__device__ __forceinline__ unsigned short f2bf(float f) {
    unsigned int u = __builtin_bit_cast(unsigned int, f);
    u += 0x7FFFu + ((u >> 16) & 1u);           // round-to-nearest-even
    return (unsigned short)(u >> 16);
}
__device__ __forceinline__ float bf2f(unsigned short b) {
    unsigned int u = ((unsigned int)b) << 16;
    return __builtin_bit_cast(float, u);
}

// A-fragment (16-bit, 16x32 MxK): per lane two contiguous 16B runs.
// base = row_ptr + kt*32 + (lane>=16 ? 8 : 0), must be 16B aligned.
__device__ __forceinline__ v16bf load_afrag(const unsigned short* base) {
    BFrag f;
    f.u4[0] = *(const v4u*)(base);
    f.u4[1] = *(const v4u*)(base + 16);
    return f.v;
}
__device__ __forceinline__ v16bf load_bfrag(const unsigned short* frag) {
    BFrag f;
    const v4u* src = (const v4u*)frag;
    f.u4[0] = src[0];
    f.u4[1] = src[1];
    return f.v;
}

#define C_DIM   128
#define HW      128
#define L_TOK   64
#define DIN     256
#define DSTATE  16
#define DTRANK  8

// ---- LDS layout (dynamic), sizes in bytes --------------------------------
// regionA: x tile f32[128][64] (32768) -> xc bf16[64][272] -> y bf16[64][272]
//          (scan gates y over xc IN PLACE: thread d owns column d)
#define SZ_REGION_A  (64 * 272 * 2)          // 34816
#define OFF_A        0
// tn (LayerNorm out) bf16 [64][136]
#define SZ_TN        (64 * 136 * 2)          // 17408
#define OFF_TN       (OFF_A + SZ_REGION_A)
// xz bf16 [64][520] (cols 0..255 = xi, 256..511 = z)
//   overlaid after scan by out f32[128][68] (34816 B)
#define SZ_XZ        (64 * 520 * 2)          // 66560
#define OFF_XZ       (OFF_TN + SZ_TN)
// dbl f32 [64][48]   (dr 0..7, B 8..23, C 24..39, 40..47 pad)
#define SZ_DBL       (64 * 48 * 4)           // 12288
#define OFF_DBL      (OFF_XZ + SZ_XZ)
#define SMEM_BYTES   (OFF_DBL + SZ_DBL)      // 131072 B = 128 KB exactly

// ws layout (ushort elements): packed bf16 B-fragments, 512 elts (1KB) each
#define WSOFF_WIN    0                        // 128 frags (4 kt x 32 nt)
#define WSOFF_WOUT   (128 * 512)              // 64 frags  (8 kt x 8 nt)
#define WSOFF_WX     (WSOFF_WOUT + 64 * 512)  // 24 frags  (8 kt x 3 nt)

// ---------------------------------------------------------------------------
// Pack kernel: convert W_in (128x512), W_out (256x128), W_x (256x40, N padded
// to 48 with zeros) into bf16 B-matrix fragments.
// B layout (16-bit, 32x16 KxN, wave32):
//   lane 0-15 : K = e (e<8), 8+e (e>=8),  N = lane
//   lane 16-31: K = 8+e (e<8), 16+e (e>=8), N = lane-16
// Fragment f stored as frag[f*512 + lane*16 + (2v+h)].
// ---------------------------------------------------------------------------
__global__ __launch_bounds__(32) void wmamba_pack_kernel(
    const float* __restrict__ W_in, const float* __restrict__ W_out,
    const float* __restrict__ W_x, unsigned short* __restrict__ ws) {
    int f    = blockIdx.x;
    int lane = threadIdx.x;
    int lN   = lane & 15;
    int kgrp = (lane >> 4) ? 8 : 0;
    if (f < 128) {                 // W_in: kt = f>>5 (K32), nt = f&31 (N16)
        int kt = f >> 5, nt = f & 31;
        int N  = nt * 16 + lN;
        unsigned short* dst = ws + WSOFF_WIN + f * 512 + lane * 16;
        #pragma unroll
        for (int e = 0; e < 16; ++e)
            dst[e] = f2bf(W_in[(kt * 32 + kgrp + e + (e >> 3) * 8) * 512 + N]);
    } else if (f < 192) {          // W_out: kt = g>>3, nt = g&7
        int g = f - 128;
        int kt = g >> 3, nt = g & 7;
        int N  = nt * 16 + lN;
        unsigned short* dst = ws + WSOFF_WOUT + g * 512 + lane * 16;
        #pragma unroll
        for (int e = 0; e < 16; ++e)
            dst[e] = f2bf(W_out[(kt * 32 + kgrp + e + (e >> 3) * 8) * 128 + N]);
    } else {                       // W_x: g = kt*3 + nt, N padded to 48
        int g = f - 192;
        int kt = g / 3, nt = g % 3;
        int N  = nt * 16 + lN;
        unsigned short* dst = ws + WSOFF_WX + g * 512 + lane * 16;
        #pragma unroll
        for (int e = 0; e < 16; ++e)
            dst[e] = (N < 40) ? f2bf(W_x[(kt * 32 + kgrp + e + (e >> 3) * 8) * 40 + N])
                              : (unsigned short)0;
    }
}

// ---------------------------------------------------------------------------
// Main fused kernel: one 256-thread block (8 waves) per 8x8 window.
// ---------------------------------------------------------------------------
__global__ __launch_bounds__(256, 2) void wmamba_main_kernel(
    const float* __restrict__ x,
    const float* __restrict__ ln_g, const float* __restrict__ ln_b,
    const float* __restrict__ b_in,
    const float* __restrict__ conv_w, const float* __restrict__ conv_b,
    const float* __restrict__ dt_w, const float* __restrict__ dt_b,
    const float* __restrict__ A_log, const float* __restrict__ Dv,
    const float* __restrict__ b_out,
    const unsigned short* __restrict__ wsfrag,
    float* __restrict__ out) {

    extern __shared__ char smem[];
    float*          s_x   = (float*)(smem + OFF_A);           // [128][64]
    unsigned short* s_xcb = (unsigned short*)(smem + OFF_A);  // [64][272] bf16 (overlay)
    unsigned short* s_y   = (unsigned short*)(smem + OFF_A);  // [64][272] bf16 (in-place)
    unsigned short* s_tn  = (unsigned short*)(smem + OFF_TN); // [64][136] bf16
    unsigned short* s_xz  = (unsigned short*)(smem + OFF_XZ); // [64][520] bf16
    float*          s_out = (float*)(smem + OFF_XZ);          // [128][68] (overlay)
    float*          s_dbl = (float*)(smem + OFF_DBL);         // [64][48]

    const int tid  = threadIdx.x;
    const int lane = tid & 31;
    const int wave = tid >> 5;
    const int lN   = lane & 15;
    const int lhi  = lane >> 4;

    const int blk = blockIdx.x;            // n = 4*16*16 = 1024 windows
    const int ww  = blk & 15;
    const int wh  = (blk >> 4) & 15;
    const int bb  = blk >> 8;

    const unsigned short* fin  = wsfrag + WSOFF_WIN;
    const unsigned short* fout = wsfrag + WSOFF_WOUT;
    const unsigned short* fwx  = wsfrag + WSOFF_WX;

    // ---- Stage 0: stage x tile (ch-major) into LDS; prefetch weights ------
    __builtin_prefetch(fin + tid * 256, 0, 0);    // packed W_in (128 KB)
    __builtin_prefetch(fout + tid * 128, 0, 0);   // packed W_out (64 KB)
    #pragma unroll
    for (int q = 0; q < 4; ++q) {
        int row = tid + q * 256;                   // 0..1023 = (ch, r)
        int ch = row >> 3, r = row & 7;
        const float* src = x + ((bb * C_DIM + ch) * HW + wh * 8 + r) * HW + ww * 8;
        float4 a0 = *(const float4*)src;
        float4 a1 = *(const float4*)(src + 4);
        *(float4*)&s_x[ch * 64 + r * 8]     = a0;
        *(float4*)&s_x[ch * 64 + r * 8 + 4] = a1;
    }
    __syncthreads();

    // ---- Stage 1: LayerNorm over C per token -> tn (bf16) -----------------
    if (tid < L_TOK) {
        int t = tid;
        float s = 0.f, s2 = 0.f;
        for (int c = 0; c < C_DIM; ++c) {
            float v = s_x[c * 64 + t];
            s += v; s2 += v * v;
        }
        float mu   = s * (1.0f / C_DIM);
        float var  = s2 * (1.0f / C_DIM) - mu * mu;
        float rstd = rsqrtf(var + 1e-5f);
        for (int c = 0; c < C_DIM; ++c) {
            float v = (s_x[c * 64 + t] - mu) * rstd * ln_g[c] + ln_b[c];
            s_tn[t * 136 + c] = f2bf(v);
        }
    }
    __syncthreads();

    // ---- Stage 2: GEMM1  tn(64x128) @ W_in(128x512) -> xz (bf16) ----------
    for (int j = 0; j < 4; ++j) {
        int nt = wave * 4 + j;
        v16bf bw[4];
        #pragma unroll
        for (int kt = 0; kt < 4; ++kt)
            bw[kt] = load_bfrag(fin + ((kt * 32 + nt) * 512 + lane * 16));
        float bin = b_in[nt * 16 + lN];
        #pragma unroll
        for (int mt = 0; mt < 4; ++mt) {
            v8f acc = {0.f, 0.f, 0.f, 0.f, 0.f, 0.f, 0.f, 0.f};
            const unsigned short* arow = s_tn + (mt * 16 + lN) * 136 + lhi * 8;
            #pragma unroll
            for (int kt = 0; kt < 4; ++kt)
                acc = __builtin_amdgcn_wmma_f32_16x16x32_bf16(
                    false, load_afrag(arow + kt * 32), false, bw[kt],
                    (short)0, acc, false, false);
            #pragma unroll
            for (int i = 0; i < 8; ++i) {
                int M = mt * 16 + i + lhi * 8;
                s_xz[M * 520 + nt * 16 + lN] = f2bf(acc[i] + bin);
            }
        }
    }
    __syncthreads();

    // ---- Stage 3: causal depthwise conv (K=4) + SiLU -> xc (bf16) ---------
    {
        int d = tid;
        float cw0 = conv_w[d * 4 + 0], cw1 = conv_w[d * 4 + 1];
        float cw2 = conv_w[d * 4 + 2], cw3 = conv_w[d * 4 + 3];
        float cb  = conv_b[d];
        float h0 = 0.f, h1 = 0.f, h2 = 0.f;   // xi[t-3], xi[t-2], xi[t-1]
        for (int t = 0; t < L_TOK; ++t) {
            float xt = bf2f(s_xz[t * 520 + d]);
            float a  = cb + h0 * cw0 + h1 * cw1 + h2 * cw2 + xt * cw3;
            h0 = h1; h1 = h2; h2 = xt;
            float v = a / (1.f + __expf(-a));          // silu
            s_xcb[t * 272 + d] = f2bf(v);              // A-matrix for GEMM2 + scan x_t
        }
    }
    __syncthreads();

    // ---- Stage 4: GEMM2  xc(64x256) @ W_x(256x48pad) -> dbl (WMMA) --------
    // 12 tiles (4 mt x 3 nt) over 8 waves; cols 40..47 land in s_dbl pad.
    #pragma unroll
    for (int pass = 0; pass < 2; ++pass) {
        int tl = wave + pass * 8;
        if (tl < 12) {
            int mt = tl & 3, nt = tl >> 2;
            v8f acc = {0.f, 0.f, 0.f, 0.f, 0.f, 0.f, 0.f, 0.f};
            const unsigned short* arow = s_xcb + (mt * 16 + lN) * 272 + lhi * 8;
            #pragma unroll
            for (int kt = 0; kt < 8; ++kt)
                acc = __builtin_amdgcn_wmma_f32_16x16x32_bf16(
                    false, load_afrag(arow + kt * 32),
                    false, load_bfrag(fwx + ((kt * 3 + nt) * 512 + lane * 16)),
                    (short)0, acc, false, false);
            int col = nt * 16 + lN;                    // 0..47
            #pragma unroll
            for (int i = 0; i < 8; ++i)
                s_dbl[(mt * 16 + i + lhi * 8) * 48 + col] = acc[i];
        }
    }
    __syncthreads();

    // ---- Stage 5: selective scan, thread d owns channel d -----------------
    // Reads xc (bf16) at (t,d) and overwrites the same slot with gated y:
    // race-free since only thread d touches column d.
    {
        int d = tid;
        float Arow[DSTATE];
        #pragma unroll
        for (int s = 0; s < DSTATE; ++s) Arow[s] = -__expf(A_log[d * DSTATE + s]);
        float dtw[DTRANK];
        #pragma unroll
        for (int r = 0; r < DTRANK; ++r) dtw[r] = dt_w[r * DIN + d];
        float dtb = dt_b[d];
        float Dd  = Dv[d];
        float h[DSTATE];
        #pragma unroll
        for (int s = 0; s < DSTATE; ++s) h[s] = 0.f;

        for (int t = 0; t < L_TOK; ++t) {
            float dr = dtb;
            #pragma unroll
            for (int r = 0; r < DTRANK; ++r) dr += s_dbl[t * 48 + r] * dtw[r];
            float delta = dr;
            if (dr < 20.f) delta = __logf(1.f + __expf(dr));   // softplus
            float xt = bf2f(s_xcb[t * 272 + d]);
            float du = delta * xt;
            float y = 0.f;
            #pragma unroll
            for (int s = 0; s < DSTATE; ++s) {
                float dA = __expf(delta * Arow[s]);
                h[s] = dA * h[s] + du * s_dbl[t * 48 + 8 + s];
                y += h[s] * s_dbl[t * 48 + 24 + s];
            }
            y += xt * Dd;
            float z = bf2f(s_xz[t * 520 + 256 + d]);
            float g = z / (1.f + __expf(-z));                  // silu(z)
            s_y[t * 272 + d] = f2bf(y * g);                    // in-place over xc
        }
    }
    __syncthreads();

    // ---- Stage 6: GEMM3  y(64x256) @ W_out(256x128) -> s_out (ch-major) ---
    {
        int nt = wave;                            // 8 N-tiles, one per wave
        v16bf bw[8];
        #pragma unroll
        for (int kt = 0; kt < 8; ++kt)
            bw[kt] = load_bfrag(fout + ((kt * 8 + nt) * 512 + lane * 16));
        #pragma unroll
        for (int mt = 0; mt < 4; ++mt) {
            v8f acc = {0.f, 0.f, 0.f, 0.f, 0.f, 0.f, 0.f, 0.f};
            const unsigned short* arow = s_y + (mt * 16 + lN) * 272 + lhi * 8;
            #pragma unroll
            for (int kt = 0; kt < 8; ++kt)
                acc = __builtin_amdgcn_wmma_f32_16x16x32_bf16(
                    false, load_afrag(arow + kt * 32), false, bw[kt],
                    (short)0, acc, false, false);
            int ch = nt * 16 + lN;
            #pragma unroll
            for (int i = 0; i < 8; ++i)
                s_out[ch * 68 + mt * 16 + i + lhi * 8] = acc[i];
        }
    }
    __syncthreads();

    // ---- Stage 7: bias + residual + coalesced store -----------------------
    #pragma unroll
    for (int q = 0; q < 4; ++q) {
        int row = tid + q * 256;                  // (ch, r)
        int ch = row >> 3, r = row & 7;
        size_t off = (size_t)((bb * C_DIM + ch) * HW + wh * 8 + r) * HW + ww * 8;
        const float* px = x + off;
        float* po = out + off;
        float bo = b_out[ch];
        float4 r0 = *(const float4*)px;
        float4 r1 = *(const float4*)(px + 4);
        float4 v0 = *(float4*)&s_out[ch * 68 + r * 8];
        float4 v1 = *(float4*)&s_out[ch * 68 + r * 8 + 4];
        v0.x += bo + r0.x; v0.y += bo + r0.y; v0.z += bo + r0.z; v0.w += bo + r0.w;
        v1.x += bo + r1.x; v1.y += bo + r1.y; v1.z += bo + r1.z; v1.w += bo + r1.w;
        *(float4*)po       = v0;
        *(float4*)(po + 4) = v1;
    }
}

// ---------------------------------------------------------------------------
extern "C" void kernel_launch(void* const* d_in, const int* in_sizes, int n_in,
                              void* d_out, int out_size, void* d_ws, size_t ws_size,
                              hipStream_t stream) {
    (void)in_sizes; (void)n_in; (void)out_size; (void)ws_size;
    const float* x      = (const float*)d_in[0];
    const float* ln_g   = (const float*)d_in[1];
    const float* ln_b   = (const float*)d_in[2];
    const float* W_in   = (const float*)d_in[3];
    const float* b_in   = (const float*)d_in[4];
    const float* conv_w = (const float*)d_in[5];
    const float* conv_b = (const float*)d_in[6];
    const float* W_x    = (const float*)d_in[7];
    const float* dt_w   = (const float*)d_in[8];
    const float* dt_b   = (const float*)d_in[9];
    const float* A_log  = (const float*)d_in[10];
    const float* Dv     = (const float*)d_in[11];
    const float* W_out  = (const float*)d_in[12];
    const float* b_out  = (const float*)d_in[13];
    float* out          = (float*)d_out;
    unsigned short* ws  = (unsigned short*)d_ws;

    // Pack all weights to bf16 WMMA-B fragments (216 fragments, 32 lanes each).
    wmamba_pack_kernel<<<dim3(216), dim3(32), 0, stream>>>(W_in, W_out, W_x, ws);

    // Allow > default dynamic LDS (128 KB; 2 workgroups fit per 320 KB WGP).
    (void)hipFuncSetAttribute((const void*)wmamba_main_kernel,
                              hipFuncAttributeMaxDynamicSharedMemorySize,
                              SMEM_BYTES);

    wmamba_main_kernel<<<dim3(1024), dim3(256), SMEM_BYTES, stream>>>(
        x, ln_g, ln_b, b_in, conv_w, conv_b, dt_w, dt_b, A_log, Dv, b_out,
        ws, out);
}
